// SimpleRNN_3839700762779
// MI455X (gfx1250) — matrix operand order repaired and checked by reading, not compiled, not run on hardware
//
#include <hip/hip_runtime.h>
#include <hip/hip_bf16.h>

// ---------------------------------------------------------------------------
// SimpleRNN (relu), B=64 T=512 I=128 H=1024 L=4 O=128, on gfx1250 (CDNA5).
//  - All GEMMs via V_WMMA_F32_16X16X32_BF16 (fp32 accumulate, bf16 operands).
//  - Input projections: time-parallel tiled GEMM (M=32768), async LDS fills.
//  - Recurrence: persistent kernel, 16 WGs, atomic grid barrier per timestep.
//    W_hh slice (128KB) + full h state (128KB) resident in 320KB WGP LDS:
//    inner k-loop is pure ds_load + v_wmma, no barriers, no global traffic.
// ---------------------------------------------------------------------------

#define RB 64
#define RT 512
#define RI 128
#define RH 1024
#define RL 4
#define RO 128

typedef __attribute__((ext_vector_type(16))) __bf16 v16bf;
typedef __attribute__((ext_vector_type(8)))  float  v8f;
typedef int vec4i __attribute__((vector_size(16)));   // b128 payload type

// ---- CDNA5 async global->LDS path (guarded; falls back to ld/st) -----------
#if defined(__gfx1250__) && __has_builtin(__builtin_amdgcn_global_load_async_to_lds_b128)
#define HAVE_ASYNC_LDS 1
#else
#define HAVE_ASYNC_LDS 0
#endif

__device__ __forceinline__ void lds_copy16(void* lds_dst, const void* g_src) {
#if HAVE_ASYNC_LDS
  __builtin_amdgcn_global_load_async_to_lds_b128(
      (__attribute__((address_space(1))) vec4i*)g_src,
      (__attribute__((address_space(3))) vec4i*)lds_dst, 0, 0);
#else
  *(uint4*)lds_dst = *(const uint4*)g_src;
#endif
}

__device__ __forceinline__ void lds_copy_wait() {
#if HAVE_ASYNC_LDS
#if __has_builtin(__builtin_amdgcn_s_wait_asynccnt)
  __builtin_amdgcn_s_wait_asynccnt(0);
#else
  asm volatile("s_wait_asynccnt 0x0" ::: "memory");
#endif
#endif
}

// ---- workspace layout (bytes; every section is 256-aligned) ----------------
static constexpr size_t OFF_BAR  = 0;
static constexpr size_t OFF_XBF  = 256;                                        // x  bf16  [B,T,I]
static constexpr size_t OFF_WIH0 = OFF_XBF  + (size_t)RB*RT*RI*2;              // Wih0 bf16 [H,I]
static constexpr size_t OFF_WHH  = OFF_WIH0 + (size_t)RH*RI*2;                 // Whh bf16 [4,H,H]
static constexpr size_t OFF_WIH  = OFF_WHH  + (size_t)4*RH*RH*2;               // Wih bf16 [3,H,H]
static constexpr size_t OFF_WFC  = OFF_WIH  + (size_t)3*RH*RH*2;               // Wfc bf16 [O,H]
static constexpr size_t OFF_XIN  = OFF_WFC  + (size_t)RO*RH*2;                 // xin f32  [B,T,H]
static constexpr size_t OFF_YS   = OFF_XIN  + (size_t)RB*RT*RH*4;              // ys  bf16 [B,T,H]
static constexpr size_t OFF_HBF  = OFF_YS   + (size_t)RB*RT*RH*2;              // h   bf16 [B,H]

// ---- helpers ----------------------------------------------------------------
__device__ __forceinline__ unsigned short f2bf(float f) {
  unsigned int u = __float_as_uint(f);
  unsigned int r = (u + 0x7FFFu + ((u >> 16) & 1u)) >> 16;   // round-nearest-even
  return (unsigned short)r;
}

// A fragment: 16x32 bf16, rows [row_base,row_base+16) of LDS tile, row stride `ld`.
// lanes 0-15 row M=lane, K 0..15; lanes 16-31 K 16..31 (matches B fragment).
__device__ __forceinline__ v16bf load_a_frag(const unsigned short* smem, int ld,
                                             int row_base, int lane) {
  const int half = lane >> 4;
  const int row  = row_base + (lane & 15);
  const __bf16* p = (const __bf16*)(smem + row * ld);
  v16bf a;
#pragma unroll
  for (int i = 0; i < 16; ++i) a[i] = p[half * 16 + i];
  return a;
}

// B fragment: 32x16 bf16 (K x N), LDS tile stored [n][k] (row-major W, x @ W^T).
// Same k order as A (not ISA table order): lane n=lane&15, lanes0-15 hold K 0..15, lanes16-31 K 16..31.
__device__ __forceinline__ v16bf load_b_frag(const unsigned short* smem, int ld,
                                             int n_base, int lane) {
  const int half = lane >> 4;
  const int n    = n_base + (lane & 15);
  const __bf16* p = (const __bf16*)(smem + n * ld);
  v16bf b;
#pragma unroll
  for (int i = 0; i < 16; ++i) b[i] = p[half * 16 + i];
  return b;
}

__device__ __forceinline__ v8f wmma_bf16(v16bf a, v16bf b, v8f c) {
  return __builtin_amdgcn_wmma_f32_16x16x32_bf16(
      /*neg_a=*/false, a, /*neg_b=*/false, b,
      /*c_mod=*/(short)0, c, /*reuse_a=*/false, /*reuse_b=*/false);
}

__device__ __forceinline__ void grid_sync(unsigned int* cnt, unsigned int nblocks,
                                          unsigned int* iter) {
  __syncthreads();
  if (threadIdx.x == 0) {
    __threadfence();  // make h-state stores visible device-wide
    const unsigned int target = (*iter + 1u) * nblocks;
    __hip_atomic_fetch_add(cnt, 1u, __ATOMIC_ACQ_REL, __HIP_MEMORY_SCOPE_AGENT);
    while (__hip_atomic_load(cnt, __ATOMIC_ACQUIRE, __HIP_MEMORY_SCOPE_AGENT) < target) {
      __builtin_amdgcn_s_sleep(1);
    }
  }
  __syncthreads();
  ++*iter;
}

// ---- kernels ----------------------------------------------------------------
__global__ void f32_to_bf16_kernel(const float* __restrict__ src,
                                   unsigned short* __restrict__ dst, int n) {
  int i = blockIdx.x * blockDim.x + threadIdx.x;
  if (i < n) dst[i] = f2bf(src[i]);
}

__global__ void zero_u32_kernel(unsigned int* p) {
  if (threadIdx.x == 0 && blockIdx.x == 0) *p = 0u;
}

// C[M,N](f32) = A[M,K](bf16) @ Bw[N,K](bf16)^T + bias0 + bias1
// block tile 128x64, 8 waves (4x2), 2x2 WMMA tiles per wave, BK=32, LDS-staged.
__global__ __launch_bounds__(256) void wmma_gemm_bias(
    const unsigned short* __restrict__ A, const unsigned short* __restrict__ Bw,
    const float* __restrict__ bias0, const float* __restrict__ bias1,
    float* __restrict__ C, int M, int N, int K) {
  __shared__ unsigned short Asm[128 * 32];
  __shared__ unsigned short Bsm[64 * 32];

  const int tid  = threadIdx.x;
  const int lane = tid & 31;
  const int wid  = tid >> 5;
  const int wm   = wid & 3;    // 4 row groups of 32
  const int wn   = wid >> 2;   // 2 col groups of 32
  const int m0   = blockIdx.y * 128;
  const int n0   = blockIdx.x * 64;

  const v8f vzero = {0.f, 0.f, 0.f, 0.f, 0.f, 0.f, 0.f, 0.f};
  v8f acc[2][2];
#pragma unroll
  for (int i = 0; i < 2; ++i)
#pragma unroll
    for (int j = 0; j < 2; ++j) acc[i][j] = vzero;

  for (int k0 = 0; k0 < K; k0 += 32) {
    __syncthreads();
    // A tile: 128x32 bf16 = 512 x 16B segments, 2 per thread (async to LDS).
#pragma unroll
    for (int s = 0; s < 2; ++s) {
      const int seg = tid + s * 256;
      const int row = seg >> 2;
      const int off = (seg & 3) << 3;
      lds_copy16(&Asm[row * 32 + off], &A[(size_t)(m0 + row) * K + k0 + off]);
    }
    // B tile: 64x32 bf16 = 256 x 16B segments, 1 per thread.
    {
      const int row = tid >> 2;
      const int off = (tid & 3) << 3;
      lds_copy16(&Bsm[row * 32 + off], &Bw[(size_t)(n0 + row) * K + k0 + off]);
      if (k0 + 32 < K) {  // pull next weight K-slice (global_prefetch_b8)
        __builtin_prefetch(&Bw[(size_t)(n0 + row) * K + k0 + 32 + off], 0, 1);
      }
    }
    lds_copy_wait();
    __syncthreads();

    v16bf af0 = load_a_frag(Asm, 32, wm * 32, lane);
    v16bf af1 = load_a_frag(Asm, 32, wm * 32 + 16, lane);
    v16bf bf0 = load_b_frag(Bsm, 32, wn * 32, lane);
    v16bf bf1 = load_b_frag(Bsm, 32, wn * 32 + 16, lane);
    acc[0][0] = wmma_bf16(af0, bf0, acc[0][0]);
    acc[0][1] = wmma_bf16(af0, bf1, acc[0][1]);
    acc[1][0] = wmma_bf16(af1, bf0, acc[1][0]);
    acc[1][1] = wmma_bf16(af1, bf1, acc[1][1]);
  }

  // epilogue: C/D layout -> VGPR v: lanes0-15 M=v, lanes16-31 M=8+v; N=lane&15
  const int half = lane >> 4;
  const int nl   = lane & 15;
#pragma unroll
  for (int i = 0; i < 2; ++i) {
#pragma unroll
    for (int j = 0; j < 2; ++j) {
      const int mg = m0 + wm * 32 + i * 16 + half * 8;
      const int ng = n0 + wn * 32 + j * 16 + nl;
      const float bsum = (bias0 ? bias0[ng] : 0.f) + (bias1 ? bias1[ng] : 0.f);
#pragma unroll
      for (int v = 0; v < 8; ++v) {
        C[(size_t)(mg + v) * N + ng] = acc[i][j][v] + bsum;
      }
    }
  }
}

// ---- persistent recurrence --------------------------------------------------
// LDS row stride: 1024+8 elems = 2064 B = 516 dwords == 4 banks shift per row,
// so 16 consecutive rows hit 16 distinct banks on fragment loads.
#define LDSW 1032
static constexpr size_t RNN_LDS_BYTES = (size_t)2 * 64 * LDSW * 2;  // ~258 KB

__global__ __launch_bounds__(256) void rnn_recur(
    const float* __restrict__ xin,            // [B,T,H] f32
    const unsigned short* __restrict__ Whh,   // [H,H] bf16
    const float* __restrict__ h0,             // [B,H] f32
    unsigned short* __restrict__ hbf,         // [B,H] bf16 state (global exchange)
    unsigned short* __restrict__ ys,          // [B,T,H] bf16 layer output
    float* __restrict__ hn,                   // [B,H] f32 final hidden
    unsigned int* __restrict__ bar, int T) {
  extern __shared__ unsigned short smem[];
  unsigned short* Wsm = smem;                  // [64][LDSW]  W_hh slice, resident
  unsigned short* Hsm = smem + 64 * LDSW;      // [64][LDSW]  full h, per timestep

  const int tid  = threadIdx.x;
  const int lane = tid & 31;
  const int wid  = tid >> 5;
  const int wm   = wid & 3;    // 4 row groups of 16 (covers B=64)
  const int wn   = wid >> 2;   // 2 col groups of 32
  const int n0   = blockIdx.x * 64;
  const unsigned int nblocks = gridDim.x;
  unsigned int iter = 0;

  // Stage the block's W_hh slice into LDS once (64 rows x 2 KB = 8192 segs/32ea).
  for (int s = tid; s < 64 * 128; s += 256) {
    const int row = s >> 7;
    const int col = (s & 127) << 3;
    lds_copy16(&Wsm[row * LDSW + col], &Whh[(size_t)(n0 + row) * RH + col]);
  }
  // initialize bf16 h state from fp32 hidden[l]
  for (int idx = blockIdx.x * 256 + tid; idx < RB * RH; idx += (int)nblocks * 256)
    hbf[idx] = f2bf(h0[idx]);
  lds_copy_wait();
  grid_sync(bar, nblocks, &iter);

  const int half = lane >> 4;
  const int nl   = lane & 15;
  const v8f vzero = {0.f, 0.f, 0.f, 0.f, 0.f, 0.f, 0.f, 0.f};

  for (int t = 0; t < T; ++t) {
    // pull the full h state into LDS (128 KB, async)
    for (int s = tid; s < 64 * 128; s += 256) {
      const int row = s >> 7;
      const int col = (s & 127) << 3;
      lds_copy16(&Hsm[row * LDSW + col], &hbf[(size_t)row * RH + col]);
    }
    lds_copy_wait();
    __syncthreads();

    // barrier-free inner loop: pure ds_load + v_wmma
    v8f acc0 = vzero, acc1 = vzero;
#pragma unroll 4
    for (int k0 = 0; k0 < RH; k0 += 32) {
      v16bf a  = load_a_frag(&Hsm[k0], LDSW, wm * 16, lane);
      v16bf b0 = load_b_frag(&Wsm[k0], LDSW, wn * 32, lane);
      v16bf b1 = load_b_frag(&Wsm[k0], LDSW, wn * 32 + 16, lane);
      acc0 = wmma_bf16(a, b0, acc0);
      acc1 = wmma_bf16(a, b1, acc1);
    }

    // epilogue: add xin, relu, emit ys (bf16), update h state, final hn (f32)
#pragma unroll
    for (int j = 0; j < 2; ++j) {
      const v8f accj = j ? acc1 : acc0;
      const int mg = wm * 16 + half * 8;
      const int ng = n0 + wn * 32 + j * 16 + nl;
#pragma unroll
      for (int v = 0; v < 8; ++v) {
        const int m = mg + v;
        const size_t bt = (size_t)m * T + t;
        float val = accj[v] + xin[bt * RH + ng];
        val = val > 0.f ? val : 0.f;
        ys[bt * RH + ng] = f2bf(val);
        hbf[(size_t)m * RH + ng] = f2bf(val);
        if (t == T - 1) hn[(size_t)m * RH + ng] = val;
      }
    }
    grid_sync(bar, nblocks, &iter);  // h visible device-wide before next step
  }
}

// ---- host orchestration -----------------------------------------------------
extern "C" void kernel_launch(void* const* d_in, const int* in_sizes, int n_in,
                              void* d_out, int out_size, void* d_ws, size_t ws_size,
                              hipStream_t stream) {
  const float* x      = (const float*)d_in[0];
  const float* hidden = (const float*)d_in[1];
  const float* Wih0   = (const float*)d_in[2];
  const float* Whh0   = (const float*)d_in[3];
  const float* bih0   = (const float*)d_in[4];
  const float* bhh0   = (const float*)d_in[5];
  const float* WihL   = (const float*)d_in[6];
  const float* WhhL   = (const float*)d_in[7];
  const float* bihL   = (const float*)d_in[8];
  const float* bhhL   = (const float*)d_in[9];
  const float* Wfc    = (const float*)d_in[10];
  const float* bfc    = (const float*)d_in[11];

  char* ws = (char*)d_ws;
  unsigned int*   bar  = (unsigned int*)(ws + OFF_BAR);
  unsigned short* xbf  = (unsigned short*)(ws + OFF_XBF);
  unsigned short* wih0 = (unsigned short*)(ws + OFF_WIH0);
  unsigned short* whh  = (unsigned short*)(ws + OFF_WHH);   // [4][H*H]
  unsigned short* wih  = (unsigned short*)(ws + OFF_WIH);   // [3][H*H]
  unsigned short* wfc  = (unsigned short*)(ws + OFF_WFC);
  float*          xin  = (float*)(ws + OFF_XIN);
  unsigned short* ys   = (unsigned short*)(ws + OFF_YS);
  unsigned short* hbf  = (unsigned short*)(ws + OFF_HBF);

  float* out = (float*)d_out;                 // [B,T,O]
  float* hn  = out + (size_t)RB * RT * RO;    // [L,B,H]

  // allow >64KB dynamic LDS for the persistent recurrence kernel
  (void)hipFuncSetAttribute((const void*)rnn_recur,
                            hipFuncAttributeMaxDynamicSharedMemorySize,
                            (int)RNN_LDS_BYTES);

  auto cvt = [&](const float* s, unsigned short* d, int n) {
    f32_to_bf16_kernel<<<(n + 255) / 256, 256, 0, stream>>>(s, d, n);
  };
  cvt(x,    xbf,                   RB * RT * RI);
  cvt(Wih0, wih0,                  RH * RI);
  cvt(Whh0, whh,                   RH * RH);
  cvt(WhhL, whh + (size_t)RH * RH, 3 * RH * RH);
  cvt(WihL, wih,                   3 * RH * RH);
  cvt(Wfc,  wfc,                   RO * RH);

  const int M = RB * RT;  // 32768

  // layer 0: input projection + recurrence
  wmma_gemm_bias<<<dim3(RH / 64, M / 128), 256, 0, stream>>>(
      xbf, wih0, bih0, bhh0, xin, M, RH, RI);
  zero_u32_kernel<<<1, 32, 0, stream>>>(bar);
  rnn_recur<<<16, 256, RNN_LDS_BYTES, stream>>>(
      xin, whh, hidden, hbf, ys, hn, bar, RT);

  // layers 1..3
  for (int l = 1; l < RL; ++l) {
    wmma_gemm_bias<<<dim3(RH / 64, M / 128), 256, 0, stream>>>(
        ys, wih + (size_t)(l - 1) * RH * RH,
        bihL + (size_t)(l - 1) * RH, bhhL + (size_t)(l - 1) * RH, xin, M, RH, RH);
    zero_u32_kernel<<<1, 32, 0, stream>>>(bar);
    rnn_recur<<<16, 256, RNN_LDS_BYTES, stream>>>(
        xin, whh + (size_t)l * RH * RH, hidden + (size_t)l * RB * RH,
        hbf, ys, hn + (size_t)l * RB * RH, bar, RT);
  }

  // output projection: [M,H] @ [O,H]^T + b_fc -> d_out
  wmma_gemm_bias<<<dim3(RO / 64, M / 128), 256, 0, stream>>>(
      ys, wfc, bfc, nullptr, out, M, RO, RH);
}
